// MultiViewGraph_25769804417
// MI455X (gfx1250) — matrix operand order repaired and checked by reading, not compiled, not run on hardware
//
#include <hip/hip_runtime.h>
#include <hip/hip_bf16.h>

typedef __attribute__((ext_vector_type(16))) __bf16 v16bf;
typedef __attribute__((ext_vector_type(8)))  __bf16 v8bf;
typedef __attribute__((ext_vector_type(8)))  float  v8f;
typedef __attribute__((ext_vector_type(4)))  unsigned int u32x4;
typedef __attribute__((ext_vector_type(8)))  int i32x8;
typedef __attribute__((ext_vector_type(4)))  int i32x4;

#define P_    576
#define B0_   64
#define NODES (B0_ * P_)   // 36864
#define KNN   9

#define HAVE_TDM __has_builtin(__builtin_amdgcn_tensor_load_to_lds)

// ---------------------------------------------------------------------------
// Tensor Data Mover: async 2-D tile (tile_k x tile_rows) global->LDS.
// D# bitfields per CDNA5 ISA ch.8: group0 {count=1, lds_addr, global_addr,
// type=2}; group1 {data_size=2B, tensor_dim0/1, tile_dim0/1, dim0_stride}.
// Tracked by TENSORcnt.
// ---------------------------------------------------------------------------
__device__ __forceinline__ void tdm_load_tile_2d(unsigned lds_byte_off,
                                                 const __bf16* gptr,
                                                 unsigned tile_k, unsigned tile_rows,
                                                 unsigned tensor_k, unsigned tensor_rows,
                                                 unsigned row_stride_elems) {
#if HAVE_TDM
  const unsigned long long ga = (unsigned long long)gptr;
  u32x4 g0;
  g0[0] = 1u;                                                  // count=1 (valid), no gather
  g0[1] = lds_byte_off;                                        // LDS dest byte address
  g0[2] = (unsigned)ga;                                        // global_addr[31:0]
  g0[3] = (unsigned)((ga >> 32) & 0x01FFFFFFu) | (2u << 30);   // global_addr[56:32] | type=2
  i32x8 g1;
  g1[0] = (int)(1u << 16);                                     // wg_mask=0, data_size=1 (2 bytes)
  g1[1] = (int)((tensor_k & 0xFFFFu) << 16);                   // tensor_dim0[15:0]
  g1[2] = (int)((tensor_k >> 16) | ((tensor_rows & 0xFFFFu) << 16));
  g1[3] = (int)((tensor_rows >> 16) | (tile_k << 16));         // tensor_dim1 hi | tile_dim0
  g1[4] = (int)tile_rows;                                      // tile_dim1 (tile_dim2=0)
  g1[5] = (int)row_stride_elems;                               // tensor_dim0_stride low32
  g1[6] = 0;
  g1[7] = 0;
  i32x4 z4 = {};
#if __clang_major__ >= 23
  i32x8 z8 = {};
  __builtin_amdgcn_tensor_load_to_lds(g0, g1, z4, z4, z8, 0);
#else
  __builtin_amdgcn_tensor_load_to_lds(g0, g1, z4, z4, 0);
#endif
#else
  (void)lds_byte_off; (void)gptr; (void)tile_k; (void)tile_rows;
  (void)tensor_k; (void)tensor_rows; (void)row_stride_elems;
#endif
}

// ---------------------------------------------------------------------------
// 1) NCHW fp32 -> (B0*P, 256) bf16 node matrix, tiled LDS transpose
// ---------------------------------------------------------------------------
__global__ __launch_bounds__(256) void k_nodes(const float* __restrict__ x,
                                               __bf16* __restrict__ node) {
  __shared__ float tile[32][33];
  const int img = blockIdx.z;
  const int p0 = blockIdx.x * 32;     // 18 tiles over P=576
  const int c0 = blockIdx.y * 32;     // 8 tiles over C=256
  const int tx = threadIdx.x & 31, ty = threadIdx.x >> 5;
  const float* xs = x + (size_t)img * 256 * P_;
#pragma unroll
  for (int r = 0; r < 4; r++)
    tile[ty + r * 8][tx] = xs[(size_t)(c0 + ty + r * 8) * P_ + (p0 + tx)];
  __syncthreads();
#pragma unroll
  for (int r = 0; r < 4; r++) {
    const int p = p0 + ty + r * 8;
    const int c = c0 + tx;
    node[((size_t)img * P_ + p) * 256 + c] = (__bf16)tile[tx][ty + r * 8];
  }
}

// ---------------------------------------------------------------------------
// 2) W_space (K x N fp32) -> Wt (N x K bf16), so B-tiles are contiguous in K
// ---------------------------------------------------------------------------
__global__ __launch_bounds__(256) void k_wt(const float* __restrict__ W,
                                            __bf16* __restrict__ Wt) {
  __shared__ float tile[32][33];
  const int k0 = blockIdx.x * 32;
  const int n0 = blockIdx.y * 32;
  const int tx = threadIdx.x & 31, ty = threadIdx.x >> 5;
#pragma unroll
  for (int r = 0; r < 4; r++)
    tile[ty + r * 8][tx] = W[(size_t)(k0 + ty + r * 8) * 256 + (n0 + tx)];
  __syncthreads();
#pragma unroll
  for (int r = 0; r < 4; r++)
    Wt[(size_t)(n0 + ty + r * 8) * 256 + (k0 + tx)] = (__bf16)tile[tx][ty + r * 8];
}

// ---------------------------------------------------------------------------
// 3) WMMA GEMM: h[36864,256] = node(bf16) @ W_space (B staged as Wt = W^T).
//    256 threads = 8 waves; 128x128 C tile; double-buffered LDS fed by TDM:
//    waves 0..3 DMA 32x32 A sub-tiles, waves 4..7 DMA 32x32 B sub-tiles,
//    tile i+1 in flight while tile i runs its 8 WMMAs per wave.
// ---------------------------------------------------------------------------
#define M_BLK 128
#define N_BLK 128
#define K_TOT 256
#define K_STEP 32
#define KITERS (K_TOT / K_STEP)

__global__ __launch_bounds__(256) void k_gemm_space(const __bf16* __restrict__ A,
                                                    const __bf16* __restrict__ Bt,
                                                    float* __restrict__ C) {
  // [buf][ A 128x32 (4096 elem) | B 128x32 (4096 elem) ] = 2 x 16 KB
  __shared__ alignas(128) __bf16 smem[2][8192];
  const int t = threadIdx.x;
  const int wave = t >> 5, lane = t & 31;
  const int m0 = blockIdx.x * M_BLK;
  const int n0 = blockIdx.y * N_BLK;
  const int wm = (wave >> 1) * 32;          // 4 M-waves
  const int wn = (wave & 1) * 64;           // 2 N-waves
  const int fm = lane & 15;
  const int ka  = (lane >> 4) * 16;         // A frag K base (contiguous K, matches B)
  const int kbB = (lane >> 4) * 16;         // B frag K base
  v8f acc[2][4] = {};

#if HAVE_TDM
  // Escape the LDS buffer into an asm with a memory clobber: the TDM writes
  // LDS behind the compiler's back, and without this the fragment ds_loads
  // are provably-unwritten memory and get folded to undef/zero.
  asm volatile("" : : "r"((const void*)&smem[0][0]) : "memory");
  auto issue = [&](int k0, int buf) {
    const unsigned base = (unsigned)buf * 16384u;   // byte offset of buffer
    if (wave < 4)
      tdm_load_tile_2d(base + (unsigned)wave * 2048u,
                       A + (size_t)(m0 + wave * 32) * K_TOT + k0,
                       K_STEP, 32, K_TOT, 1u << 20, K_TOT);
    else
      tdm_load_tile_2d(base + 8192u + (unsigned)(wave - 4) * 2048u,
                       Bt + (size_t)(n0 + (wave - 4) * 32) * K_TOT + k0,
                       K_STEP, 32, K_TOT, 1u << 20, K_TOT);
  };
  issue(0, 0);
#endif

  for (int it = 0; it < KITERS; it++) {
    const int buf = it & 1;
    const __bf16* sAb = &smem[buf][0];
    const __bf16* sBb = &smem[buf][4096];

#if HAVE_TDM
    __builtin_amdgcn_s_wait_tensorcnt(0);   // my sub-tile of tile `it` landed
    __syncthreads();                        // everyone's sub-tiles landed
    asm volatile("" : : : "memory");        // tile is now readable: force real ds_loads
    if (it + 1 < KITERS) issue((it + 1) * K_STEP, buf ^ 1);  // overlap next DMA
#else
    // fallback: synchronous VGPR staging (same LDS layout)
    {
      const int k0 = it * K_STEP;
      const int rowc = t >> 1, half = t & 1;
      *(uint4*)&smem[buf][rowc * 32 + half * 16] =
          *(const uint4*)(A + (size_t)(m0 + rowc) * K_TOT + k0 + half * 16);
      *(uint4*)&smem[buf][4096 + rowc * 32 + half * 16] =
          *(const uint4*)(Bt + (size_t)(n0 + rowc) * K_TOT + k0 + half * 16);
    }
    __syncthreads();
#endif

    v16bf af[2], bf4[4];
#pragma unroll
    for (int i = 0; i < 2; i++) {
      const __bf16* ap = sAb + (wm + i * 16 + fm) * 32;
      v8bf lo = *(const v8bf*)(ap + ka);
      v8bf hi = *(const v8bf*)(ap + ka + 8);
      af[i] = __builtin_shufflevector(lo, hi, 0, 1, 2, 3, 4, 5, 6, 7,
                                      8, 9, 10, 11, 12, 13, 14, 15);
    }
#pragma unroll
    for (int j = 0; j < 4; j++)
      bf4[j] = *(const v16bf*)(sBb + (wn + j * 16 + fm) * 32 + kbB);
#pragma unroll
    for (int i = 0; i < 2; i++)
#pragma unroll
      for (int j = 0; j < 4; j++)
        acc[i][j] = __builtin_amdgcn_wmma_f32_16x16x32_bf16(
            false, af[i], false, bf4[j], (short)0, acc[i][j], false, false);
  }

#pragma unroll
  for (int i = 0; i < 2; i++) {
    const int rbase = m0 + wm + i * 16 + (lane >> 4) * 8;
#pragma unroll
    for (int j = 0; j < 4; j++) {
      const int col = n0 + wn + j * 16 + fm;
#pragma unroll
      for (int r = 0; r < 8; r++)
        C[(size_t)(rbase + r) * 256 + col] = acc[i][j][r];
    }
  }
}

// ---------------------------------------------------------------------------
// 4) Per-(node,head) attention logits: aS = <h, att_src>, aD = <h, att_dst>
// ---------------------------------------------------------------------------
__global__ __launch_bounds__(256) void k_logits(const float* __restrict__ h,
                                                const float* __restrict__ ws,
                                                const float* __restrict__ wd,
                                                float* __restrict__ aS,
                                                float* __restrict__ aD, int npair) {
  const int wave = (int)((blockIdx.x * blockDim.x + threadIdx.x) >> 5);
  const int lane = threadIdx.x & 31;
  if (wave >= npair) return;
  const int n = wave >> 1, head = wave & 1;
  const float* hp = h + (size_t)n * 256 + head * 128;
  const float* as = ws + head * 128;
  const float* ad = wd + head * 128;
  float s = 0.f, d = 0.f;
#pragma unroll
  for (int j = 0; j < 4; j++) {
    const float v = hp[lane + j * 32];
    s += v * as[lane + j * 32];
    d += v * ad[lane + j * 32];
  }
#pragma unroll
  for (int off = 16; off; off >>= 1) {
    s += __shfl_xor(s, off);
    d += __shfl_xor(d, off);
  }
  if (lane == 0) { aS[n * 2 + head] = s; aD[n * 2 + head] = d; }
}

// ---------------------------------------------------------------------------
// 5) Space aggregation: softmax over 9 KNN srcs + self, weighted gather of h.
//    x_space written straight into d_out (final reshape is a reinterpret).
// ---------------------------------------------------------------------------
__global__ __launch_bounds__(256) void k_space(const float* __restrict__ h,
                                               const int* __restrict__ es,
                                               const float* __restrict__ aS,
                                               const float* __restrict__ aD,
                                               const float* __restrict__ bias,
                                               float* __restrict__ out) {
  const int n = blockIdx.x;
  __shared__ int srcs[10];
  __shared__ float alpha[2][10];
  const int t = threadIdx.x;
  if (t < 10) srcs[t] = (t < KNN) ? es[n * KNN + t] : n;   // dst[e]==n by construction
  __syncthreads();
  if (t < 20) {
    const int head = t / 10, k = t % 10;
    float e = aS[srcs[k] * 2 + head] + aD[n * 2 + head];
    alpha[head][k] = (e > 0.f) ? e : 0.2f * e;             // leaky_relu(0.2)
  }
  __syncthreads();
  if (t < 2) {
    float m = -1e30f;
    for (int k = 0; k < 10; k++) m = fmaxf(m, alpha[t][k]);
    float ssum = 0.f;
    for (int k = 0; k < 10; k++) { float e = __expf(alpha[t][k] - m); alpha[t][k] = e; ssum += e; }
    const float inv = 1.f / (ssum + 1e-16f);
    for (int k = 0; k < 10; k++) alpha[t][k] *= inv;
  }
  __syncthreads();
  const int head = t >> 7;
  float acc = 0.f;
#pragma unroll
  for (int k = 0; k < 10; k++)
    acc += alpha[head][k] * h[(size_t)srcs[k] * 256 + t];  // L2-resident gathers
  out[(size_t)n * 256 + t] = acc + bias[t];
}

// ---------------------------------------------------------------------------
// 6) Per-image mean pool over P=576 patches
// ---------------------------------------------------------------------------
__global__ __launch_bounds__(256) void k_emb(const float* __restrict__ xs,
                                             float* __restrict__ emb) {
  const int img = blockIdx.x, c = threadIdx.x;
  const float* p = xs + (size_t)img * P_ * 256 + c;
  float s = 0.f;
  for (int i = 0; i < P_; i++) s += p[(size_t)i * 256];
  emb[img * 256 + c] = s * (1.0f / (float)P_);
}

// ---------------------------------------------------------------------------
// 7) View linear: hv[i] = emb[i%64] @ W_view  (tiny)
// ---------------------------------------------------------------------------
__global__ __launch_bounds__(256) void k_viewlin(const float* __restrict__ emb,
                                                 const float* __restrict__ Wv,
                                                 float* __restrict__ hv) {
  const int i = blockIdx.x, c = threadIdx.x;
  const float* e = emb + (i & 63) * 256;
  float s = 0.f;
  for (int k = 0; k < 256; k++) s += e[k] * Wv[k * 256 + c];
  hv[i * 256 + c] = s;
}

// 7b) View logits: one wave per node, 256 channels
__global__ __launch_bounds__(256) void k_vlogits(const float* __restrict__ hv,
                                                 const float* __restrict__ ws,
                                                 const float* __restrict__ wd,
                                                 float* __restrict__ aSv,
                                                 float* __restrict__ aDv) {
  const int wave = (int)((blockIdx.x * blockDim.x + threadIdx.x) >> 5);
  const int lane = threadIdx.x & 31;
  if (wave >= 128) return;
  const float* hp = hv + wave * 256;
  float s = 0.f, d = 0.f;
#pragma unroll
  for (int j = 0; j < 8; j++) {
    const float v = hp[lane + 32 * j];
    s += v * ws[lane + 32 * j];
    d += v * wd[lane + 32 * j];
  }
#pragma unroll
  for (int off = 16; off; off >>= 1) {
    s += __shfl_xor(s, off);
    d += __shfl_xor(d, off);
  }
  if (lane == 0) { aSv[wave] = s; aDv[wave] = d; }
}

// ---------------------------------------------------------------------------
// 8) View aggregation: serial deterministic edge scan + 256-thread gather
// ---------------------------------------------------------------------------
__global__ __launch_bounds__(256) void k_vagg(const float* __restrict__ hv,
                                              const int* __restrict__ ev, int E,
                                              const float* __restrict__ aSv,
                                              const float* __restrict__ aDv,
                                              const float* __restrict__ bias,
                                              float* __restrict__ feat) {
  const int i = blockIdx.x;
  __shared__ int srcs[64];
  __shared__ float alpha[64];
  __shared__ int cnt_s;
  const int t = threadIdx.x;
  if (t == 0) {
    int cnt = 0;
    for (int e = 0; e < E; e++)
      if (ev[E + e] == i && cnt < 63) srcs[cnt++] = ev[e];
    srcs[cnt++] = i;  // self loop
    float m = -1e30f;
    for (int k = 0; k < cnt; k++) {
      float v = aSv[srcs[k]] + aDv[i];
      v = (v > 0.f) ? v : 0.2f * v;
      alpha[k] = v;
      m = fmaxf(m, v);
    }
    float ssum = 0.f;
    for (int k = 0; k < cnt; k++) { float e = __expf(alpha[k] - m); alpha[k] = e; ssum += e; }
    const float inv = 1.f / (ssum + 1e-16f);
    for (int k = 0; k < cnt; k++) alpha[k] *= inv;
    cnt_s = cnt;
  }
  __syncthreads();
  const int cnt = cnt_s;
  float acc = 0.f;
  for (int k = 0; k < cnt; k++) acc += alpha[k] * hv[srcs[k] * 256 + t];
  feat[i * 256 + t] = acc + bias[t];
}

// ---------------------------------------------------------------------------
// 9) Final blend in place: out = 0.5*x_space + 0.5*feat[img]
// ---------------------------------------------------------------------------
__global__ __launch_bounds__(256) void k_final(float* __restrict__ out,
                                               const float* __restrict__ feat) {
  const int n = blockIdx.x;
  const size_t idx = (size_t)n * 256 + threadIdx.x;
  const int img = n / P_;
  out[idx] = 0.5f * out[idx] + 0.5f * feat[img * 256 + threadIdx.x];
}

// ---------------------------------------------------------------------------
extern "C" void kernel_launch(void* const* d_in, const int* in_sizes, int n_in,
                              void* d_out, int out_size, void* d_ws, size_t ws_size,
                              hipStream_t stream) {
  (void)n_in; (void)out_size; (void)ws_size;
  const float* x   = (const float*)d_in[0];
  const int*   es  = (const int*)d_in[3];
  const int*   ev  = (const int*)d_in[4];
  const float* Wsp = (const float*)d_in[5];
  const float* asp = (const float*)d_in[6];
  const float* adp = (const float*)d_in[7];
  const float* bsp = (const float*)d_in[8];
  const float* Wv  = (const float*)d_in[9];
  const float* asv = (const float*)d_in[10];
  const float* adv = (const float*)d_in[11];
  const float* bv  = (const float*)d_in[12];
  float* out = (float*)d_out;
  const int Ev = in_sizes[4] / 2;

  char* ws = (char*)d_ws;
  size_t off = 0;
  auto take = [&](size_t bytes) -> void* {
    void* p = ws + off;
    off = (off + bytes + 255) & ~(size_t)255;
    return p;
  };
  __bf16* nodeb = (__bf16*)take((size_t)NODES * 256 * 2);
  __bf16* WtS   = (__bf16*)take((size_t)256 * 256 * 2);
  float* h    = (float*)take((size_t)NODES * 256 * 4);
  float* aS   = (float*)take((size_t)NODES * 2 * 4);
  float* aD   = (float*)take((size_t)NODES * 2 * 4);
  float* emb  = (float*)take((size_t)64 * 256 * 4);
  float* hv   = (float*)take((size_t)128 * 256 * 4);
  float* aSv  = (float*)take((size_t)128 * 4);
  float* aDv  = (float*)take((size_t)128 * 4);
  float* feat = (float*)take((size_t)128 * 256 * 4);

  k_nodes<<<dim3(P_ / 32, 256 / 32, B0_), 256, 0, stream>>>(x, nodeb);
  k_wt<<<dim3(8, 8), 256, 0, stream>>>(Wsp, WtS);
  k_gemm_space<<<dim3(NODES / M_BLK, 256 / N_BLK), 256, 0, stream>>>(nodeb, WtS, h);
  k_logits<<<(NODES * 2 + 7) / 8, 256, 0, stream>>>(h, asp, adp, aS, aD, NODES * 2);
  k_space<<<NODES, 256, 0, stream>>>(h, es, aS, aD, bsp, out);
  k_emb<<<B0_, 256, 0, stream>>>(out, emb);
  k_viewlin<<<128, 256, 0, stream>>>(emb, Wv, hv);
  k_vlogits<<<16, 256, 0, stream>>>(hv, asv, adv, aSv, aDv);
  k_vagg<<<128, 256, 0, stream>>>(hv, ev, Ev, aSv, aDv, bv, feat);
  k_final<<<NODES, 256, 0, stream>>>(out, feat);
}
